// ResidualBottleneck_88974542504303
// MI455X (gfx1250) — matrix operand order repaired and checked by reading, not compiled, not run on hardware
//
#include <hip/hip_runtime.h>
#include <hip/hip_bf16.h>

#define NUM_Q 8
#define B_ 8
#define T_ 2048
#define D_ 512
#define K_ 1024
#define RB 514  // bf16 LDS row stride (even, 257 dwords -> conflict-free)

typedef __attribute__((ext_vector_type(16))) __bf16 v16bf;
typedef __attribute__((ext_vector_type(8)))  float  v8f;
typedef __attribute__((ext_vector_type(8)))  unsigned int v8u;

__device__ __forceinline__ unsigned short f2bf(float f) {
  unsigned int u = __builtin_bit_cast(unsigned int, f);
  unsigned int r = (u + 0x7fffu + ((u >> 16) & 1u)) >> 16;  // RNE
  return (unsigned short)r;
}

__device__ __forceinline__ v16bf load_bfrag(const unsigned short* bptr, int kk, int hi) {
  v8u bu;
#pragma unroll
  for (int p = 0; p < 8; ++p) {
    const int kb = kk + ((p & 3) * 2) + ((p >> 2) * 16) + hi * 8;     // B layout: VGPR p -> K = {2p (p<4) | 16+2(p-4)} + 8*hi
    bu[p] = *(const unsigned int*)&bptr[kb]; // two 16B runs/lane -> b128 pairs
  }
  return __builtin_bit_cast(v16bf, bu);
}

// ---- zero the per-stage commit accumulators (capture-safe) ----
__global__ void vq_zero(float* s) {
  if (threadIdx.x < NUM_Q) s[threadIdx.x] = 0.0f;
}

// ---- codebook fp32 -> bf16, plus 0.5*||c||^2 ----
__global__ __launch_bounds__(128)
void vq_prep(const float* __restrict__ cb, unsigned short* __restrict__ cbh,
             float* __restrict__ nrmh) {
  const int cw  = blockIdx.x;        // 0..8191  (q*1024 + k)
  const int tid = threadIdx.x;       // 128 threads
  const float* src = cb + (size_t)cw * D_;
  unsigned short* dst = cbh + (size_t)cw * D_;
  float ss = 0.0f;
#pragma unroll
  for (int j = 0; j < 4; ++j) {
    float f = src[tid + 128 * j];
    dst[tid + 128 * j] = f2bf(f);
    ss += f * f;
  }
#pragma unroll
  for (int off = 16; off >= 1; off >>= 1) ss += __shfl_xor(ss, off, 32);
  __shared__ float sred[4];
  if ((tid & 31) == 0) sred[tid >> 5] = ss;
  __syncthreads();
  if (tid == 0) nrmh[cw] = 0.5f * (sred[0] + sred[1] + sred[2] + sred[3]);
}

// ---- fused 8-stage RVQ ----
__global__ __launch_bounds__(256)
void vq_main(const float* __restrict__ x, const float* __restrict__ cb,
             const unsigned short* __restrict__ cbh,
             const float* __restrict__ nrmh,
             float* __restrict__ out, float* __restrict__ stage_sums) {
  __shared__ unsigned short s_res[32 * RB];   // bf16 residual tile
  __shared__ float s_cval[32][4];
  __shared__ int   s_cidx[32][4];
  __shared__ int   s_idx[32];
  __shared__ float s_wsum[8];

  const int tid  = threadIdx.x;
  const int lane = tid & 31;
  const int wave = tid >> 5;
  const int blk  = blockIdx.x;            // 512 blocks
  const int b    = blk >> 6;
  const int t0   = (blk & 63) << 5;       // 32 t-rows per block

  const int hi      = lane >> 4;
  const int col     = lane & 15;
  const int m_tile  = wave & 1;
  const int n_group = wave >> 1;

  // residual in registers: element (row = lane, d = wave*64 + i)
  float res[64];
  const size_t xbase = ((size_t)b * D_) * T_ + (size_t)t0 + (size_t)lane;
  const int d0 = wave * 64;
#pragma unroll
  for (int i = 0; i < 64; ++i)
    res[i] = x[xbase + (size_t)(d0 + i) * T_];   // coalesced along t

  const size_t N = (size_t)B_ * D_ * T_;

  for (int q = 0; q < NUM_Q; ++q) {
    // ---- phase A: registers -> bf16 LDS (A-operand source) ----
#pragma unroll
    for (int i = 0; i < 64; ++i)
      s_res[lane * RB + d0 + i] = f2bf(res[i]);
    __syncthreads();

    // ---- phase B: WMMA distance GEMM + per-wave argmin ----
    const unsigned short* cbq = cbh + (size_t)q * K_ * D_;
    const float* nq = nrmh + q * K_;
    float bval[8];
    int   bidx[8];
#pragma unroll
    for (int j = 0; j < 8; ++j) { bval[j] = 3.4e38f; bidx[j] = 0; }

    const int arow = m_tile * 16 + col;
    // 4 n-tiles in flight per wave -> 4 independent WMMA chains, A reused 4x
    for (int ntg = 0; ntg < 4; ++ntg) {
      const int c0 = n_group * 256 + ntg * 64 + col;    // lane's codeword per tile
      const unsigned short* bp0 = cbq + (size_t)(c0)      * D_;
      const unsigned short* bp1 = cbq + (size_t)(c0 + 16) * D_;
      const unsigned short* bp2 = cbq + (size_t)(c0 + 32) * D_;
      const unsigned short* bp3 = cbq + (size_t)(c0 + 48) * D_;
      // prefetch the half-norms so scoring doesn't wait on a fresh load
      const float nh0 = nq[c0], nh1 = nq[c0 + 16], nh2 = nq[c0 + 32], nh3 = nq[c0 + 48];
      v8f acc0 = {}, acc1 = {}, acc2 = {}, acc3 = {};
#pragma unroll 2
      for (int kk = 0; kk < 512; kk += 32) {
        v8u au;
#pragma unroll
        for (int p = 0; p < 8; ++p) {
          // A layout: VGPR p -> K = {2p (p<4) | 16+2(p-4)} + 8*hi
          const int ka = kk + ((p & 3) * 2) + ((p >> 2) * 16) + hi * 8;
          au[p] = *(const unsigned int*)&s_res[arow * RB + ka];
        }
        const v16bf a = __builtin_bit_cast(v16bf, au);
        acc0 = __builtin_amdgcn_wmma_f32_16x16x32_bf16(
            false, a, false, load_bfrag(bp0, kk, hi), (short)0, acc0, false, false);
        acc1 = __builtin_amdgcn_wmma_f32_16x16x32_bf16(
            false, a, false, load_bfrag(bp1, kk, hi), (short)0, acc1, false, false);
        acc2 = __builtin_amdgcn_wmma_f32_16x16x32_bf16(
            false, a, false, load_bfrag(bp2, kk, hi), (short)0, acc2, false, false);
        acc3 = __builtin_amdgcn_wmma_f32_16x16x32_bf16(
            false, a, false, load_bfrag(bp3, kk, hi), (short)0, acc3, false, false);
      }
#pragma unroll
      for (int j = 0; j < 8; ++j) {
        float sc;
        sc = nh0 - acc0[j]; if (sc < bval[j]) { bval[j] = sc; bidx[j] = c0; }
        sc = nh1 - acc1[j]; if (sc < bval[j]) { bval[j] = sc; bidx[j] = c0 + 16; }
        sc = nh2 - acc2[j]; if (sc < bval[j]) { bval[j] = sc; bidx[j] = c0 + 32; }
        sc = nh3 - acc3[j]; if (sc < bval[j]) { bval[j] = sc; bidx[j] = c0 + 48; }
      }
    }
    // reduce over the 16 column-lanes (wave32 halves)
#pragma unroll
    for (int j = 0; j < 8; ++j) {
      float v = bval[j]; int ix = bidx[j];
#pragma unroll
      for (int off = 8; off >= 1; off >>= 1) {
        const float ov = __shfl_xor(v, off, 16);
        const int   oi = __shfl_xor(ix, off, 16);
        if (ov < v || (ov == v && oi < ix)) { v = ov; ix = oi; }
      }
      bval[j] = v; bidx[j] = ix;
    }
    if (col == 0) {
#pragma unroll
      for (int j = 0; j < 8; ++j) {
        const int row = m_tile * 16 + j + 8 * hi;
        s_cval[row][n_group] = bval[j];
        s_cidx[row][n_group] = bidx[j];
      }
    }
    __syncthreads();
    // final argmin across the 4 N-groups (1 thread per row)
    if (tid < 32) {
      float v = s_cval[tid][0]; int ix = s_cidx[tid][0];
#pragma unroll
      for (int g = 1; g < 4; ++g) {
        const float ov = s_cval[tid][g]; const int oi = s_cidx[tid][g];
        if (ov < v || (ov == v && oi < ix)) { v = ov; ix = oi; }
      }
      s_idx[tid] = ix;
    }
    __syncthreads();

    // ---- phase D: gather q (contiguous 256B/lane), update residual, outputs ----
    const float* qrow = cb + ((size_t)q * K_ + (size_t)s_idx[lane]) * D_ + d0;
    float* qo = (q < 2) ? (out + (size_t)(q + 1) * N) : nullptr;
    float ss = 0.0f;
#pragma unroll
    for (int i = 0; i < 64; ++i) {
      const float qv = qrow[i];               // vectorizable b128 gathers
      const float r  = res[i] - qv;
      res[i] = r;
      ss += r * r;                            // (q - residual)^2 == r_new^2
      if (q < 2) qo[xbase + (size_t)(d0 + i) * T_] = qv;  // q_st == q exactly
    }
#pragma unroll
    for (int off = 16; off >= 1; off >>= 1) ss += __shfl_xor(ss, off, 32);
    if (lane == 0) s_wsum[wave] = ss;
    __syncthreads();
    if (tid == 0) {
      float tot = 0.0f;
#pragma unroll
      for (int w = 0; w < 8; ++w) tot += s_wsum[w];
      atomicAdd(&stage_sums[q], tot);
    }
    __syncthreads();
  }

  // out = sum(q_i) = x - residual_final  (coalesced along t)
#pragma unroll
  for (int i = 0; i < 64; ++i)
    out[xbase + (size_t)(d0 + i) * T_] = x[xbase + (size_t)(d0 + i) * T_] - res[i];
}

// ---- epilogue: com = mean over stages of 0.25*sum/N ----
__global__ void vq_final(const float* __restrict__ s, float* __restrict__ com) {
  float t = 0.0f;
#pragma unroll
  for (int i = 0; i < NUM_Q; ++i) t += s[i];
  const float invN = 1.0f / (float)((size_t)B_ * T_ * D_);
  *com = t * 0.25f * invN * (1.0f / (float)NUM_Q);
}

extern "C" void kernel_launch(void* const* d_in, const int* in_sizes, int n_in,
                              void* d_out, int out_size, void* d_ws, size_t ws_size,
                              hipStream_t stream) {
  (void)in_sizes; (void)n_in; (void)out_size; (void)ws_size;
  const float* x  = (const float*)d_in[0];   // [8, 512, 2048] f32
  const float* cb = (const float*)d_in[1];   // [8, 1024, 512] f32
  float* out = (float*)d_out;                // out | q1 | q2 | com (flat)

  unsigned short* cbh = (unsigned short*)d_ws;                       // 8 MB bf16 codebooks
  float* nrmh = (float*)((char*)d_ws + (size_t)NUM_Q * K_ * D_ * 2); // 32 KB half-norms
  float* sums = nrmh + NUM_Q * K_;                                   // 8 stage accumulators

  vq_zero <<<1, 32, 0, stream>>>(sums);
  vq_prep <<<NUM_Q * K_, 128, 0, stream>>>(cb, cbh, nrmh);
  vq_main <<<(B_ * T_) / 32, 256, 0, stream>>>(x, cb, cbh, nrmh, out, sums);
  vq_final<<<1, 1, 0, stream>>>(sums, out + (size_t)3 * B_ * D_ * T_);
}
